// GAT_8040178778181
// MI455X (gfx1250) — matrix operand order repaired and checked by reading, not compiled, not run on hardware
//
#include <hip/hip_runtime.h>
#include <hip/hip_bf16.h>
#include <math.h>

#define NNODES   100000
#define EDGES    1600000
#define EATOT    (EDGES + NNODES)   // edges + self loops
#define F_IN     500
#define KPAD1    512
#define HID      16
#define HEADS    8
#define HC       (HEADS * HID)      // 128
#define NCLS     10
#define NEG_SLOPE 0.2f
#define BN_EPS    1e-5f

typedef __attribute__((ext_vector_type(16))) __bf16 v16bf;
typedef __attribute__((ext_vector_type(8)))  __bf16 v8bf;
typedef __attribute__((ext_vector_type(8)))  float  v8f;

#define CDIV(a, b) (((a) + (b) - 1) / (b))

// ---------------------------------------------------------------------------
// fp32 -> bf16 conversion / padding kernels
// ---------------------------------------------------------------------------
__global__ void k_cvt_x(const float* __restrict__ x, __bf16* __restrict__ xb) {
    int t = blockIdx.x * blockDim.x + threadIdx.x;
    if (t >= NNODES * KPAD1) return;
    int n = t >> 9;          // / 512
    int k = t & (KPAD1 - 1); // % 512
    xb[t] = (k < F_IN) ? (__bf16)x[(size_t)n * F_IN + k] : (__bf16)0.0f;
}

// W[K x Ncol] row-major -> Wt[Npad x Kpad] (per-output-column contiguous K)
__global__ void k_cvt_wt(const float* __restrict__ W, __bf16* __restrict__ Wt,
                         int K, int Kpad, int Ncol, int Npad) {
    int t = blockIdx.x * blockDim.x + threadIdx.x;
    if (t >= Npad * Kpad) return;
    int n = t / Kpad;
    int k = t - n * Kpad;
    Wt[t] = (k < K && n < Ncol) ? (__bf16)W[(size_t)k * Ncol + n] : (__bf16)0.0f;
}

// ---------------------------------------------------------------------------
// WMMA bf16 GEMM:  C[M x Ncols] = A[M x KPAD] * B[KPAD x Ncols]
//   A row-major (lda = KPAD); B pre-transposed Wt[col*KPAD + k], staged in LDS.
//   One wave computes a 16 x (NT*16) strip: A fragment loaded once per k-step,
//   NT independent WMMAs issued against LDS-resident B fragments.
// ---------------------------------------------------------------------------
template <int NT, int KPAD>
__global__ __launch_bounds__(256) void
k_wmma_gemm(const __bf16* __restrict__ A, const __bf16* __restrict__ Bt,
            float* __restrict__ C, int ldc, int M, int Ncols) {
    __shared__ __bf16 Bs[NT * 16 * KPAD];

    // cooperative copy of the whole transposed weight matrix into LDS
    {
        const uint4* src = (const uint4*)Bt;   // 16B = 8 bf16
        uint4*       dst = (uint4*)Bs;
        const int    nvec = (NT * 16 * KPAD) / 8;
        for (int i = threadIdx.x; i < nvec; i += blockDim.x) dst[i] = src[i];
    }
    __syncthreads();

    const int lane = threadIdx.x & 31;
    const int wave = blockIdx.x * (blockDim.x >> 5) + (threadIdx.x >> 5);
    const int ntm  = M >> 4;                 // M is a multiple of 16
    if (wave >= ntm) return;                 // wave-uniform: EXEC stays full
    const int row0 = wave << 4;
    const int hi   = lane >> 4;              // 0: lanes 0-15, 1: lanes 16-31
    const int l15  = lane & 15;

    // A 16x32 bf16 layout: lane(l15) row M=row0+l15; 16 contiguous K
    // kb..kb+15, kb = k0 + 16*hi.
    const __bf16* arow  = A  + (size_t)(row0 + l15) * KPAD + (hi << 4);
    // B 32x16 bf16 layout: lane(l15) col, 16 contiguous K at k0+16*hi (from LDS)
    const __bf16* bbase = Bs + l15 * KPAD + (hi << 4);

    v8f acc[NT];
#pragma unroll
    for (int nt = 0; nt < NT; ++nt) acc[nt] = v8f{0.f,0.f,0.f,0.f,0.f,0.f,0.f,0.f};

#pragma unroll 2
    for (int k0 = 0; k0 < KPAD; k0 += 32) {
        v8bf alo = *(const v8bf*)(arow + k0);
        v8bf ahi = *(const v8bf*)(arow + k0 + 8);
        v16bf a  = __builtin_shufflevector(alo, ahi,
                       0, 1, 2, 3, 4, 5, 6, 7, 8, 9, 10, 11, 12, 13, 14, 15);
        __builtin_prefetch(arow + k0 + 64, 0, 0);   // global_prefetch_b8
#pragma unroll
        for (int nt = 0; nt < NT; ++nt) {
            v16bf b = *(const v16bf*)(bbase + nt * 16 * KPAD + k0); // ds_load_b128 x2
            acc[nt] = __builtin_amdgcn_wmma_f32_16x16x32_bf16(
                          false, a, false, b, (short)0, acc[nt], false, false);
        }
    }

    // C/D layout: VGPR v -> row row0 + 8*hi + v, col nt*16 + l15
#pragma unroll
    for (int nt = 0; nt < NT; ++nt) {
        const int col = (nt << 4) + l15;
        if (col < Ncols) {
            float* crow = C + (size_t)(row0 + (hi << 3)) * ldc + col;
#pragma unroll
            for (int v = 0; v < 8; ++v) crow[(size_t)v * ldc] = acc[nt][v];
        }
    }
}

// ---------------------------------------------------------------------------
// attention logits:  al_src[n,h] = sum_c h[n,h,c] * a_src[h,c]  (same for dst)
// ---------------------------------------------------------------------------
template <int H, int C>
__global__ void k_att_logits(const float* __restrict__ Hf,
                             const float* __restrict__ a_src,
                             const float* __restrict__ a_dst,
                             float* __restrict__ als, float* __restrict__ ald) {
    int t = blockIdx.x * blockDim.x + threadIdx.x;
    if (t >= NNODES * H) return;
    int n = t / H, h = t - (t / H) * H;
    const float* hp = Hf + (size_t)n * (H * C) + h * C;
    float ss = 0.f, sd = 0.f;
#pragma unroll
    for (int c = 0; c < C; ++c) {
        float v = hp[c];
        ss += v * a_src[h * C + c];
        sd += v * a_dst[h * C + c];
    }
    als[t] = ss;
    ald[t] = sd;
}

// ---------------------------------------------------------------------------
// fills / bias init
// ---------------------------------------------------------------------------
__global__ void k_fill_u32(unsigned* __restrict__ p, unsigned v, int n) {
    int t = blockIdx.x * blockDim.x + threadIdx.x;
    if (t < n) p[t] = v;
}
__global__ void k_fill_f32(float* __restrict__ p, float v, int n) {
    int t = blockIdx.x * blockDim.x + threadIdx.x;
    if (t < n) p[t] = v;
}
__global__ void k_init_bias(float* __restrict__ agg, const float* __restrict__ b,
                            int n, int cols) {
    int t = blockIdx.x * blockDim.x + threadIdx.x;
    if (t < n) agg[t] = b[t % cols];
}

// ---------------------------------------------------------------------------
// sortable float <-> unsigned (monotone order-preserving bijection)
// ---------------------------------------------------------------------------
__device__ __forceinline__ unsigned fflip(float f) {
    unsigned u = __float_as_uint(f);
    return u ^ ((u >> 31) ? 0xFFFFFFFFu : 0x80000000u);
}
__device__ __forceinline__ float funflip(unsigned u) {
    return __uint_as_float(u ^ ((u >> 31) ? 0x80000000u : 0xFFFFFFFFu));
}

__device__ __forceinline__ void edge_sd(int e, const int* es, const int* ed,
                                        int& s, int& d) {
    if (e < EDGES) { s = es[e]; d = ed[e]; }
    else           { s = e - EDGES; d = e - EDGES; }
}

// ---------------------------------------------------------------------------
// edge pass 1: segment max of leakyrelu(al_src[s]+al_dst[d])  (hw atomic_max_u32)
// ---------------------------------------------------------------------------
template <int H>
__global__ void k_edge_max(const int* __restrict__ es, const int* __restrict__ ed,
                           const float* __restrict__ als, const float* __restrict__ ald,
                           unsigned* __restrict__ mflip) {
    int t = blockIdx.x * blockDim.x + threadIdx.x;
    if (t >= EATOT * H) return;
    int e = t / H, h = t - (t / H) * H;
    int s, d; edge_sd(e, es, ed, s, d);
    float v = als[s * H + h] + ald[d * H + h];
    v = v > 0.f ? v : NEG_SLOPE * v;
    atomicMax(&mflip[d * H + h], fflip(v));
}

// edge pass 2: denom[d,h] += exp(e - m[d,h])   (hw global_atomic_add_f32)
template <int H>
__global__ void k_edge_sum(const int* __restrict__ es, const int* __restrict__ ed,
                           const float* __restrict__ als, const float* __restrict__ ald,
                           const unsigned* __restrict__ mflip,
                           float* __restrict__ denom) {
    int t = blockIdx.x * blockDim.x + threadIdx.x;
    if (t >= EATOT * H) return;
    int e = t / H, h = t - (t / H) * H;
    int s, d; edge_sd(e, es, ed, s, d);
    float v = als[s * H + h] + ald[d * H + h];
    v = v > 0.f ? v : NEG_SLOPE * v;
    float m = funflip(mflip[d * H + h]);
    atomicAdd(&denom[d * H + h], __expf(v - m));
}

// edge pass 3: agg[d,h,:] += alpha * h[s,h,:]
template <int H, int C>
__global__ void k_edge_scatter(const int* __restrict__ es, const int* __restrict__ ed,
                               const float* __restrict__ als, const float* __restrict__ ald,
                               const unsigned* __restrict__ mflip,
                               const float* __restrict__ denom,
                               const float* __restrict__ Hf,
                               float* __restrict__ agg) {
    int t = blockIdx.x * blockDim.x + threadIdx.x;
    if (t >= EATOT * H) return;
    int e = t / H, h = t - (t / H) * H;
    int s, d; edge_sd(e, es, ed, s, d);
    float v = als[s * H + h] + ald[d * H + h];
    v = v > 0.f ? v : NEG_SLOPE * v;
    float m     = funflip(mflip[d * H + h]);
    float p     = __expf(v - m);
    float alpha = p / fmaxf(denom[d * H + h], 1e-16f);
    const float* hs = Hf  + (size_t)s * (H * C) + h * C;
    float*       ag = agg + (size_t)d * (H * C) + h * C;
#pragma unroll
    for (int c = 0; c < C; ++c) atomicAdd(&ag[c], hs[c] * alpha);
}

// ---------------------------------------------------------------------------
// ELU + eval-mode BN, emit bf16 for the next layer's WMMA GEMM
// ---------------------------------------------------------------------------
__global__ void k_elubn_bf16(const float* __restrict__ agg,
                             const float* __restrict__ g,
                             const float* __restrict__ be,
                             __bf16* __restrict__ out) {
    int t = blockIdx.x * blockDim.x + threadIdx.x;
    if (t >= NNODES * HC) return;
    int j = t & (HC - 1);
    float x = agg[t];
    float e = x > 0.f ? x : expm1f(x);
    out[t] = (__bf16)(e * (g[j] * rsqrtf(1.f + BN_EPS)) + be[j]);
}

// ---------------------------------------------------------------------------
// final log_softmax over 10 classes
// ---------------------------------------------------------------------------
__global__ void k_logsoftmax(const float* __restrict__ agg, float* __restrict__ out) {
    int n = blockIdx.x * blockDim.x + threadIdx.x;
    if (n >= NNODES) return;
    float v[NCLS];
    float mx = -1e30f;
#pragma unroll
    for (int i = 0; i < NCLS; ++i) { v[i] = agg[(size_t)n * NCLS + i]; mx = fmaxf(mx, v[i]); }
    float s = 0.f;
#pragma unroll
    for (int i = 0; i < NCLS; ++i) s += __expf(v[i] - mx);
    float ls = mx + __logf(s);
#pragma unroll
    for (int i = 0; i < NCLS; ++i) out[(size_t)n * NCLS + i] = v[i] - ls;
}

// ---------------------------------------------------------------------------
// host launch
// ---------------------------------------------------------------------------
extern "C" void kernel_launch(void* const* d_in, const int* in_sizes, int n_in,
                              void* d_out, int out_size, void* d_ws, size_t ws_size,
                              hipStream_t stream) {
    const float* x   = (const float*)d_in[0];
    const int*   ei  = (const int*)d_in[1];
    const int*   es  = ei;
    const int*   ed  = ei + EDGES;
    const float* W1  = (const float*)d_in[2];
    const float* a1s = (const float*)d_in[3];
    const float* a1d = (const float*)d_in[4];
    const float* b1  = (const float*)d_in[5];
    const float* g1  = (const float*)d_in[6];
    const float* be1 = (const float*)d_in[7];
    const float* W2  = (const float*)d_in[8];
    const float* a2s = (const float*)d_in[9];
    const float* a2d = (const float*)d_in[10];
    const float* b2  = (const float*)d_in[11];
    const float* g2  = (const float*)d_in[12];
    const float* be2 = (const float*)d_in[13];
    const float* W3  = (const float*)d_in[14];
    const float* a3s = (const float*)d_in[15];
    const float* a3d = (const float*)d_in[16];
    const float* b3  = (const float*)d_in[17];
    float* out = (float*)d_out;

    // ---- workspace bump allocator (256B aligned) ----
    uintptr_t base = (uintptr_t)d_ws;
    auto alloc = [&](size_t bytes) -> void* {
        void* p = (void*)base;
        base += (bytes + 255) & ~(size_t)255;
        return p;
    };
    __bf16*   Xb  = (__bf16*)  alloc((size_t)NNODES * KPAD1 * 2);
    __bf16*   Wt1 = (__bf16*)  alloc((size_t)HC * KPAD1 * 2);   // 128 x 512
    __bf16*   Wt2 = (__bf16*)  alloc((size_t)HC * HC * 2);      // 128 x 128
    __bf16*   Wt3 = (__bf16*)  alloc((size_t)16 * HC * 2);      // 16  x 128
    float*    Hf  = (float*)   alloc((size_t)NNODES * HC * 4);
    float*    Agg = (float*)   alloc((size_t)NNODES * HC * 4);
    __bf16*   Hb2 = (__bf16*)  alloc((size_t)NNODES * HC * 2);
    float*    ALs = (float*)   alloc((size_t)NNODES * HEADS * 4);
    float*    ALd = (float*)   alloc((size_t)NNODES * HEADS * 4);
    unsigned* Mfl = (unsigned*)alloc((size_t)NNODES * HEADS * 4);
    float*    Den = (float*)   alloc((size_t)NNODES * HEADS * 4);

    unsigned ninf_flip;
    { union { float f; unsigned u; } cv; cv.f = -1e30f; ninf_flip = cv.u ^ 0xFFFFFFFFu; }

    const int B = 256;
    const int wavesPerBlock = B / 32;
    const int ntm = NNODES >> 4;                 // 6250 strips of 16 rows
    const int edgeT8 = EATOT * HEADS;
    const int edgeT1 = EATOT;

    // ---- weight / input conversion ----
    k_cvt_x <<<CDIV(NNODES * KPAD1, B), B, 0, stream>>>(x, Xb);
    k_cvt_wt<<<CDIV(HC * KPAD1, B), B, 0, stream>>>(W1, Wt1, F_IN, KPAD1, HC, HC);
    k_cvt_wt<<<CDIV(HC * HC, B), B, 0, stream>>>(W2, Wt2, HC, HC, HC, HC);
    k_cvt_wt<<<CDIV(16 * HC, B), B, 0, stream>>>(W3, Wt3, HC, HC, NCLS, 16);

    // ================= layer 1 =================
    {
        k_wmma_gemm<8, KPAD1><<<CDIV(ntm, wavesPerBlock), B, 0, stream>>>(
            Xb, Wt1, Hf, HC, NNODES, HC);
        k_att_logits<HEADS, HID><<<CDIV(NNODES * HEADS, B), B, 0, stream>>>(
            Hf, a1s, a1d, ALs, ALd);
        k_fill_u32<<<CDIV(NNODES * HEADS, B), B, 0, stream>>>(Mfl, ninf_flip, NNODES * HEADS);
        k_fill_f32<<<CDIV(NNODES * HEADS, B), B, 0, stream>>>(Den, 0.f, NNODES * HEADS);
        k_init_bias<<<CDIV(NNODES * HC, B), B, 0, stream>>>(Agg, b1, NNODES * HC, HC);
        k_edge_max<HEADS><<<CDIV(edgeT8, B), B, 0, stream>>>(es, ed, ALs, ALd, Mfl);
        k_edge_sum<HEADS><<<CDIV(edgeT8, B), B, 0, stream>>>(es, ed, ALs, ALd, Mfl, Den);
        k_edge_scatter<HEADS, HID><<<CDIV(edgeT8, B), B, 0, stream>>>(
            es, ed, ALs, ALd, Mfl, Den, Hf, Agg);
        k_elubn_bf16<<<CDIV(NNODES * HC, B), B, 0, stream>>>(Agg, g1, be1, Hb2);
    }

    // ================= layer 2 =================
    {
        k_wmma_gemm<8, HC><<<CDIV(ntm, wavesPerBlock), B, 0, stream>>>(
            Hb2, Wt2, Hf, HC, NNODES, HC);
        k_att_logits<HEADS, HID><<<CDIV(NNODES * HEADS, B), B, 0, stream>>>(
            Hf, a2s, a2d, ALs, ALd);
        k_fill_u32<<<CDIV(NNODES * HEADS, B), B, 0, stream>>>(Mfl, ninf_flip, NNODES * HEADS);
        k_fill_f32<<<CDIV(NNODES * HEADS, B), B, 0, stream>>>(Den, 0.f, NNODES * HEADS);
        k_init_bias<<<CDIV(NNODES * HC, B), B, 0, stream>>>(Agg, b2, NNODES * HC, HC);
        k_edge_max<HEADS><<<CDIV(edgeT8, B), B, 0, stream>>>(es, ed, ALs, ALd, Mfl);
        k_edge_sum<HEADS><<<CDIV(edgeT8, B), B, 0, stream>>>(es, ed, ALs, ALd, Mfl, Den);
        k_edge_scatter<HEADS, HID><<<CDIV(edgeT8, B), B, 0, stream>>>(
            es, ed, ALs, ALd, Mfl, Den, Hf, Agg);
        k_elubn_bf16<<<CDIV(NNODES * HC, B), B, 0, stream>>>(Agg, g2, be2, Hb2);
    }

    // ================= layer 3 (1 head, 10 classes) =================
    {
        k_wmma_gemm<1, HC><<<CDIV(ntm, wavesPerBlock), B, 0, stream>>>(
            Hb2, Wt3, Hf, NCLS, NNODES, NCLS);
        k_att_logits<1, NCLS><<<CDIV(NNODES, B), B, 0, stream>>>(
            Hf, a3s, a3d, ALs, ALd);
        k_fill_u32<<<CDIV(NNODES, B), B, 0, stream>>>(Mfl, ninf_flip, NNODES);
        k_fill_f32<<<CDIV(NNODES, B), B, 0, stream>>>(Den, 0.f, NNODES);
        k_init_bias<<<CDIV(NNODES * NCLS, B), B, 0, stream>>>(Agg, b3, NNODES * NCLS, NCLS);
        k_edge_max<1><<<CDIV(edgeT1, B), B, 0, stream>>>(es, ed, ALs, ALd, Mfl);
        k_edge_sum<1><<<CDIV(edgeT1, B), B, 0, stream>>>(es, ed, ALs, ALd, Mfl, Den);
        k_edge_scatter<1, NCLS><<<CDIV(edgeT1, B), B, 0, stream>>>(
            es, ed, ALs, ALd, Mfl, Den, Hf, Agg);
        k_logsoftmax<<<CDIV(NNODES, B), B, 0, stream>>>(Agg, out);
    }
}
